// TGNModel_1279900254339
// MI455X (gfx1250) — matrix operand order repaired and checked by reading, not compiled, not run on hardware
//
#include <hip/hip_runtime.h>
#include <hip/hip_bf16.h>

typedef __attribute__((ext_vector_type(16))) _Float16 v16h;
typedef __attribute__((ext_vector_type(8)))  _Float16 v8h;
typedef __attribute__((ext_vector_type(8)))  float    v8f;

#define MEM_DIM   100
#define TIME_DIM  100
#define K1        300   // 2*MEM_DIM + TIME_DIM
#define K1P       320   // padded to 10 * 32
#define HID       100
#define HIDP      112   // 7 * 16 output col tiles
#define K2P       128   // layer-2 K padded to 4 * 32
#define OUTD      3
#define NT1       7     // layer-1 N tiles
#define NS1       10    // layer-1 K steps
#define NS2       4     // layer-2 K steps
#define MTILE     32    // edges per wave-iteration (2 x 16-row sub-tiles)
#define WPB       4     // waves per block (128 threads)
#define W1S_ELEMS (NS1*NT1*32*16)   // 35840 halfs
#define W2S_ELEMS (NS2*1*32*16)     // 2048 halfs
#define WLDS_ELEMS (W1S_ELEMS + W2S_ELEMS)          // 37888 halfs = 75776 B
#define XSLICE     (MTILE * K1P)                     // 10240 halfs per wave
#define LDS_HALFS  (WLDS_ELEMS + WPB * XSLICE)       // 78848 halfs = 157696 B
#define MAX_BLOCKS 2048

// ---------------------------------------------------------------------------
// Pre-swizzle W1 [300,100] f32 -> f16 B-fragments for v_wmma_f32_16x16x32_f16.
// Fragment layout per (k-step s, n-tile t): 32 lanes x 16 halfs contiguous.
// Lane = {hi = lane>>4, n_local = lane&15}; element i -> K = s*32 + hi*16 + i.
// ---------------------------------------------------------------------------
__global__ void swizzle_w1(const float* __restrict__ W1, _Float16* __restrict__ W1s) {
    int idx = blockIdx.x * 256 + threadIdx.x;
    if (idx >= W1S_ELEMS) return;
    int i    = idx & 15;
    int lane = (idx >> 4) & 31;
    int ft   = idx >> 9;            // s*NT1 + t
    int t    = ft % NT1;
    int s    = ft / NT1;
    int hi   = lane >> 4;
    int nl   = lane & 15;
    int k    = s * 32 + hi * 16 + i;
    int n    = t * 16 + nl;
    float v  = (k < K1 && n < HID) ? W1[k * HID + n] : 0.0f;
    W1s[idx] = (_Float16)v;
}

__global__ void swizzle_w2(const float* __restrict__ W2, _Float16* __restrict__ W2s) {
    int idx = blockIdx.x * 256 + threadIdx.x;
    if (idx >= W2S_ELEMS) return;
    int i    = idx & 15;
    int lane = (idx >> 4) & 31;
    int s    = idx >> 9;
    int hi   = lane >> 4;
    int nl   = lane & 15;
    int k    = s * 32 + hi * 16 + i;
    float v  = (k < HID && nl < OUTD) ? W2[k * OUTD + nl] : 0.0f;
    W2s[idx] = (_Float16)v;
}

// A-fragment loader: 16-bit A matrix 16x32 in contiguous k order (not the ISA table order):
// lane m(0..15)/m+16 hold row M=m; element i -> K=kbase+hi*16+i
// (matches B-frag order).  => two contiguous v8h (ds_load_b128) loads.
__device__ __forceinline__ v16h load_a_frag(const _Float16* base, int ldk,
                                            int kbase, int lane) {
    int m  = lane & 15;
    int hi = lane >> 4;
    const _Float16* p = base + m * ldk + kbase + hi * 16;
    union { v16h v; v8h h[2]; } r;
    r.h[0] = *(const v8h*)(p);
    r.h[1] = *(const v8h*)(p + 8);
    return r.v;
}

// ---------------------------------------------------------------------------
// Fused TGN edge MLP. One wave per 32-edge tile (2 sub-tiles share B-frags).
// Weights staged once per block into LDS, shared by all waves; persistent
// grid-stride over tiles.
// ---------------------------------------------------------------------------
__global__ __launch_bounds__(WPB * 32) void tgn_fused(
    const int*   __restrict__ src,
    const int*   __restrict__ dst,
    const float* __restrict__ tt,
    const float* __restrict__ memory,
    const float* __restrict__ last_update,
    const float* __restrict__ time_w,
    const float* __restrict__ time_b,
    const float* __restrict__ b1,
    const float* __restrict__ b2,
    const _Float16* __restrict__ W1s,   // pre-swizzled in workspace (W2s follows)
    float* __restrict__ out,
    int E, int n_tiles, int tile_stride)
{
    extern __shared__ __align__(16) _Float16 smem[];
    _Float16* W1lds = smem;                       // 35840 halfs
    _Float16* W2lds = smem + W1S_ELEMS;           // 2048 halfs
    const int w    = threadIdx.x >> 5;
    const int lane = threadIdx.x & 31;
    _Float16* Xw   = smem + WLDS_ELEMS + w * XSLICE;

    // ---- Stage 0: cooperative copy of swizzled weights (W1s||W2s) to LDS ----
    {
        const uint4* g = (const uint4*)W1s;       // W2s is contiguous after W1s
        uint4*       l = (uint4*)smem;
        const int nvec = (WLDS_ELEMS * 2) / 16;   // 4736 x 16B
        for (int i = threadIdx.x; i < nvec; i += WPB * 32) l[i] = g[i];
    }
    __syncthreads();

    const int nl = lane & 15;
    const int hi = lane >> 4;

    // ---- Hoisted per-lane constants (tile-invariant) ----
    // time-encode coefficients for the two col-pair chunks that can hit
    // c in [200,300): j=3 (c=192+2*lane) and j=4 (c=256+2*lane).
    float twv[2][2], tbv[2][2];
#pragma unroll
    for (int jj = 0; jj < 2; ++jj) {
#pragma unroll
        for (int p = 0; p < 2; ++p) {
            const int q = 192 + jj * 64 + 2 * lane + p - 200;
            const bool ok = (q >= 0) && (q < TIME_DIM);
            twv[jj][p] = ok ? time_w[q] : 0.0f;
            tbv[jj][p] = ok ? time_b[q] : 0.0f;
        }
    }
    float bias1[NT1];
#pragma unroll
    for (int t = 0; t < NT1; ++t) {
        const int n = t * 16 + nl;
        bias1[t] = (n < HID) ? b1[n] : 0.0f;
    }
    const float bias2 = (nl < OUTD) ? b2[nl] : 0.0f;

    // ---- Persistent loop: one 32-edge tile per wave per iteration ----
    for (int tile = blockIdx.x * WPB + w; tile < n_tiles; tile += tile_stride) {
        const int e0 = tile * MTILE;

        // ---- Stage 1: build X tile [32 x 320] f16 in LDS (zero-padded) ----
        for (int e = 0; e < MTILE; ++e) {
            const int eg = e0 + e;
            // prefetch next edge's memory rows (global_prefetch_b8)
            if (e < MTILE - 1 && (eg + 1) < E) {
                const int sn = src[eg + 1], dn = dst[eg + 1];
                const char* ps = (const char*)(memory + (long)sn * MEM_DIM);
                const char* pd = (const char*)(memory + (long)dn * MEM_DIM);
                __builtin_prefetch(ps + (lane & 3) * 128, 0, 0);
                __builtin_prefetch(pd + (lane & 3) * 128, 0, 0);
            }
            if (eg < E) {
                const int    si = src[eg];
                const int    di = dst[eg];
                const float  dt = tt[eg] - last_update[si];
                const float2* sm2 = (const float2*)(memory + (long)si * MEM_DIM);
                const float2* dm2 = (const float2*)(memory + (long)di * MEM_DIM);
#pragma unroll
                for (int j = 0; j < 5; ++j) {          // col pairs: c = j*64+2*lane
                    const int c = j * 64 + 2 * lane;
                    float x0, x1;
                    if (c < MEM_DIM) {
                        const float2 v = sm2[c >> 1];       x0 = v.x; x1 = v.y;
                    } else if (c < 2 * MEM_DIM) {
                        const float2 v = dm2[(c - MEM_DIM) >> 1]; x0 = v.x; x1 = v.y;
                    } else if (c < K1) {
                        const int jj = j - 3;               // 0 or 1
                        x0 = __cosf(dt * twv[jj][0] + tbv[jj][0]);
                        x1 = __cosf(dt * twv[jj][1] + tbv[jj][1]);
                    } else {
                        x0 = 0.0f; x1 = 0.0f;
                    }
                    union { _Float16 h[2]; unsigned u; } pk;
                    pk.h[0] = (_Float16)x0;
                    pk.h[1] = (_Float16)x1;
                    *(unsigned*)(Xw + e * K1P + c) = pk.u;  // ds_store_b32
                }
            } else {
#pragma unroll
                for (int j = 0; j < 5; ++j)
                    *(unsigned*)(Xw + e * K1P + j * 64 + 2 * lane) = 0u;
            }
        }

        // ---- Stage 2: layer 1 GEMM [32,320]x[320,112]; B-frag reused x2 ----
        v8f acc0[NT1] = {};
        v8f acc1[NT1] = {};
#pragma unroll
        for (int s = 0; s < NS1; ++s) {
            const v16h a0 = load_a_frag(Xw,              K1P, s * 32, lane);
            const v16h a1 = load_a_frag(Xw + 16 * K1P,   K1P, s * 32, lane);
#pragma unroll
            for (int t = 0; t < NT1; ++t) {
                union { v16h v; v8h h[2]; } b;
                const _Float16* bp = W1lds + ((s * NT1 + t) * 32 + lane) * 16;
                b.h[0] = *(const v8h*)(bp);
                b.h[1] = *(const v8h*)(bp + 8);
                acc0[t] = __builtin_amdgcn_wmma_f32_16x16x32_f16(
                    false, a0, false, b.v, (short)0, acc0[t], false, false);
                acc1[t] = __builtin_amdgcn_wmma_f32_16x16x32_f16(
                    false, a1, false, b.v, (short)0, acc1[t], false, false);
            }
        }

        // ---- Stage 3: bias + ReLU -> f16 H1 [32 x 128] in LDS (alias X) ----
        _Float16* Hw = Xw;                       // GEMM1 reads are done
#pragma unroll
        for (int t = 0; t < NT1; ++t) {
            const int n = t * 16 + nl;
            const float bias = bias1[t];
#pragma unroll
            for (int r = 0; r < 8; ++r) {
                const int m = r + hi * 8;
                float h0 = fmaxf(acc0[t][r] + bias, 0.0f);
                float h1 = fmaxf(acc1[t][r] + bias, 0.0f);
                Hw[m * K2P + n]          = (_Float16)h0;
                Hw[(m + 16) * K2P + n]   = (_Float16)h1;
            }
        }
        // zero pad cols 112..127 for all 32 rows (aligned 16B stores)
        {
            v8h z = {};
            *(v8h*)(Hw + nl * K2P        + HIDP + hi * 8) = z;
            *(v8h*)(Hw + (nl + 16) * K2P + HIDP + hi * 8) = z;
        }

        // ---- Stage 4: layer 2 GEMM [32,128]x[128,16]; B-frag reused x2 ----
        v8f acc2a = {};
        v8f acc2b = {};
#pragma unroll
        for (int s = 0; s < NS2; ++s) {
            const v16h a0 = load_a_frag(Hw,             K2P, s * 32, lane);
            const v16h a1 = load_a_frag(Hw + 16 * K2P,  K2P, s * 32, lane);
            union { v16h v; v8h h[2]; } b;
            const _Float16* bp = W2lds + (s * 32 + lane) * 16;
            b.h[0] = *(const v8h*)(bp);
            b.h[1] = *(const v8h*)(bp + 8);
            acc2a = __builtin_amdgcn_wmma_f32_16x16x32_f16(
                false, a0, false, b.v, (short)0, acc2a, false, false);
            acc2b = __builtin_amdgcn_wmma_f32_16x16x32_f16(
                false, a1, false, b.v, (short)0, acc2b, false, false);
        }

        // ---- Stage 5: store out[e,0..2] ----
        if (nl < OUTD) {
#pragma unroll
            for (int r = 0; r < 8; ++r) {
                const int row0 = e0 + r + hi * 8;
                const int row1 = row0 + 16;
                if (row0 < E) out[(long)row0 * OUTD + nl] = acc2a[r] + bias2;
                if (row1 < E) out[(long)row1 * OUTD + nl] = acc2b[r] + bias2;
            }
        }
    }
}

extern "C" void kernel_launch(void* const* d_in, const int* in_sizes, int n_in,
                              void* d_out, int out_size, void* d_ws, size_t ws_size,
                              hipStream_t stream) {
    const int*   src         = (const int*)  d_in[0];
    const int*   dst         = (const int*)  d_in[1];
    const float* tt          = (const float*)d_in[2];
    // d_in[3] = edge_attr : unused by the reference computation
    const float* memory      = (const float*)d_in[4];
    const float* last_update = (const float*)d_in[5];
    const float* time_w      = (const float*)d_in[6];
    const float* time_b      = (const float*)d_in[7];
    const float* W1          = (const float*)d_in[8];
    const float* b1          = (const float*)d_in[9];
    const float* W2          = (const float*)d_in[10];
    const float* b2          = (const float*)d_in[11];
    float*       out         = (float*)d_out;

    const int E = in_sizes[0];

    _Float16* W1s = (_Float16*)d_ws;
    _Float16* W2s = W1s + W1S_ELEMS;   // contiguous after W1s (copied together)

    swizzle_w1<<<(W1S_ELEMS + 255) / 256, 256, 0, stream>>>(W1, W1s);
    swizzle_w2<<<(W2S_ELEMS + 255) / 256, 256, 0, stream>>>(W2, W2s);

    const int n_tiles = (E + MTILE - 1) / MTILE;
    int blocks = (n_tiles + WPB - 1) / WPB;
    if (blocks > MAX_BLOCKS) blocks = MAX_BLOCKS;
    const int tile_stride = blocks * WPB;
    const size_t dyn_lds = (size_t)LDS_HALFS * sizeof(_Float16);  // 157696 B

    tgn_fused<<<blocks, WPB * 32, dyn_lds, stream>>>(
        src, dst, tt, memory, last_update, time_w, time_b,
        b1, b2, W1s, out, E, n_tiles, tile_stride);
}
